// SwinTransformerBlock_75900662055628
// MI455X (gfx1250) — matrix operand order repaired and checked by reading, not compiled, not run on hardware
//
#include <hip/hip_runtime.h>
#include <hip/hip_bf16.h>
#include <math.h>

// ---------------------------------------------------------------------------
// Swin Transformer block for MI455X (gfx1250), wave32 + WMMA f16 16x16x32,
// async global->LDS staging (ASYNCcnt), fused epilogues.
// B=16, C=384, H=W=56, NH=12, WS=7, SS=3, N=49 tokens/window, nW=64, HD=32.
// ---------------------------------------------------------------------------

typedef _Float16 h16;
typedef __attribute__((ext_vector_type(16))) _Float16 v16h;
typedef __attribute__((ext_vector_type(8)))  _Float16 v8h;
typedef __attribute__((ext_vector_type(8)))  float    v8f;

__device__ __forceinline__ v16h frag16(const h16* p0, const h16* p1) {
  v8h lo = *(const v8h*)p0;
  v8h hi = *(const v8h*)p1;
  return __builtin_shufflevector(lo, hi, 0,1,2,3,4,5,6,7,8,9,10,11,12,13,14,15);
}

__device__ __forceinline__ v8f wmma_f16(v16h a, v16h b, v8f c) {
  return __builtin_amdgcn_wmma_f32_16x16x32_f16(false, a, false, b, (short)0, c, false, false);
}

// --- async global -> LDS (CDNA5 VFLAT async ops, tracked by ASYNCcnt) ------
__device__ __forceinline__ void async_g2l_b128(void* lds, const void* g) {
  unsigned l = (unsigned)(size_t)lds;      // low 32 bits of generic LDS ptr
  asm volatile("global_load_async_to_lds_b128 %0, %1, off"
               :: "v"(l), "v"(g) : "memory");
}
__device__ __forceinline__ void async_g2l_b32(void* lds, const void* g) {
  unsigned l = (unsigned)(size_t)lds;
  asm volatile("global_load_async_to_lds_b32 %0, %1, off"
               :: "v"(l), "v"(g) : "memory");
}
__device__ __forceinline__ void wait_async0() {
  asm volatile("s_wait_asynccnt 0x0" ::: "memory");
}

// ---------------------------------------------------------------------------
// Weight convert + transpose: w[K][Nc] fp32 -> bt[Nc][K] f16 (K contiguous)
// ---------------------------------------------------------------------------
__global__ __launch_bounds__(256) void wcvt_kernel(const float* __restrict__ w,
                                                   h16* __restrict__ bt,
                                                   int K, int Nc) {
  int e = blockIdx.x * 256 + threadIdx.x;
  if (e >= K * Nc) return;
  int n = e / K, k = e % K;
  bt[e] = (h16)w[(size_t)k * Nc + n];
}

// ---------------------------------------------------------------------------
// LayerNorm over C=384 + (mode 0) shift(-3,-3) + window partition -> f16 rows.
// 32 tokens per block; channel-major global reads land transposed in padded
// LDS via per-lane async-to-LDS scatter.
// ---------------------------------------------------------------------------
__global__ __launch_bounds__(256) void ln_partition_kernel(
    const float* __restrict__ x, const float* __restrict__ g,
    const float* __restrict__ bta, h16* __restrict__ out, int mode) {
  __shared__ float tile[32 * 386];
  __shared__ float psum[256], psq[256];
  __shared__ float s_mean[32], s_rstd[32];
  const int tid  = threadIdx.x;
  const int tok0 = blockIdx.x * 32;          // 32 | 3136 -> block within one b
  const int b    = tok0 / 3136;
  const int hw0  = tok0 % 3136;
  const float* xb = x + (size_t)b * 384 * 3136 + hw0;
  // phase 1: coalesced channel-major loads, async scatter into LDS [t][c]
  for (int e = tid; e < 384 * 32; e += 256) {
    int c = e >> 5, t = e & 31;
    async_g2l_b32(&tile[t * 386 + c], &xb[(size_t)c * 3136 + t]);
  }
  wait_async0();
  __syncthreads();
  // phase 2: partial sums (8 chunks of 48 channels per token)
  {
    int t = tid & 31, c0 = (tid >> 5) * 48;
    float s = 0.f, q = 0.f;
    for (int c = c0; c < c0 + 48; ++c) { float v = tile[t * 386 + c]; s += v; q += v * v; }
    psum[tid] = s; psq[tid] = q;
  }
  __syncthreads();
  if (tid < 32) {
    float s = 0.f, q = 0.f;
    for (int j = 0; j < 8; ++j) { s += psum[j * 32 + tid]; q += psq[j * 32 + tid]; }
    float mu = s * (1.f / 384.f);
    float var = q * (1.f / 384.f) - mu * mu;
    s_mean[tid] = mu;
    s_rstd[tid] = rsqrtf(var + 1e-5f);
  }
  __syncthreads();
  // phase 3: normalize, f16-pack, coalesced row writes
  for (int e = tid; e < 32 * 192; e += 256) {
    int t = e / 192, cp = e % 192, c = cp * 2;
    float mu = s_mean[t], rs = s_rstd[t];
    float v0 = (tile[t * 386 + c]     - mu) * rs * g[c]     + bta[c];
    float v1 = (tile[t * 386 + c + 1] - mu) * rs * g[c + 1] + bta[c + 1];
    int token = tok0 + t;
    int hw = token % 3136;
    int row;
    if (mode == 0) {
      int h = hw / 56, w = hw % 56;
      int sh = h - 3; if (sh < 0) sh += 56;
      int sw = w - 3; if (sw < 0) sw += 56;
      int wy = sh / 7, ty = sh % 7, wx = sw / 7, tx = sw % 7;
      row = (b * 64 + wy * 8 + wx) * 49 + ty * 7 + tx;
    } else {
      row = token;
    }
    h16 h0 = (h16)v0, h1 = (h16)v1;
    unsigned u = ((unsigned)*(unsigned short*)&h1 << 16) | *(unsigned short*)&h0;
    *((unsigned*)(out + (size_t)row * 384) + cp) = u;
  }
}

// ---------------------------------------------------------------------------
// WMMA GEMM: C[row][col] = A[row][0:K] . Bt[col][0:K] + bias[col], fused epi.
// Block: 256 thr (8 waves). Block tile: 16 rows x (8*NT*16) cols.
// A tile (16 x K, contiguous) staged to LDS with async b128 copies.
// EPI 0: store f16 [row][Nc]                   (QKV)
// EPI 1: GELU(exact) then store f16 [row][Nc]  (FC1)
// EPI 2: window-reverse + unshift + x residual -> f32 (B,C,H,W)  (proj)
// EPI 3: plain token order + x2 residual       -> f32 (B,C,H,W)  (FC2)
// ---------------------------------------------------------------------------
template <int EPI, int NT>
__global__ __launch_bounds__(256) void gemm_kernel(
    const h16* __restrict__ A, const h16* __restrict__ Bt,
    const float* __restrict__ bias, void* __restrict__ outp,
    const float* __restrict__ resid, int K, int Nc) {
  extern __shared__ h16 As[];                 // 16 * K halves
  const int tid  = threadIdx.x;
  const int lane = tid & 31;
  const int wave = tid >> 5;
  const int row0 = blockIdx.x * 16;

  { // async contiguous copy of the 16xK A tile into LDS (16B per lane/op)
    const h16* src = A + (size_t)row0 * K;
    const int chunks = (16 * K) / 8;          // 16B chunks
    for (int i = tid; i < chunks; i += 256)
      async_g2l_b128((v8h*)As + i, (const v8h*)src + i);
  }
  wait_async0();
  __syncthreads();

  const int nbase = (blockIdx.y * 8 + wave) * (NT * 16);
  v8f zero = {};
  v8f acc[NT];
#pragma unroll
  for (int t = 0; t < NT; ++t) acc[t] = zero;

  const int m  = lane & 15;
  const int hs = lane >> 4;
  const h16* arow = As + m * K + hs * 8;
  const int ncol = lane & 15;

  for (int kt = 0; kt < K; kt += 32) {
    v16h af = frag16(arow + kt, arow + kt + 16);
#pragma unroll
    for (int t = 0; t < NT; ++t) {
      int cb = nbase + t * 16;
      if (cb < Nc) {                          // wave-uniform predicate
        const h16* bp = Bt + (size_t)(cb + ncol) * K + kt + hs * 8;
        __builtin_prefetch(bp + 32, 0, 0);    // next K-step of this B column
        v16h bf = frag16(bp, bp + 16);
        acc[t] = wmma_f16(af, bf, acc[t]);
      }
    }
  }

#pragma unroll
  for (int t = 0; t < NT; ++t) {
    int cb = nbase + t * 16;
    if (cb >= Nc) continue;
    int c = cb + ncol;
    float bv = bias[c];
#pragma unroll
    for (int r = 0; r < 8; ++r) {
      int row = row0 + r + hs * 8;
      float val = acc[t][r] + bv;
      if (EPI == 0) {
        ((h16*)outp)[(size_t)row * Nc + c] = (h16)val;
      } else if (EPI == 1) {
        float ge = 0.5f * val * (1.f + erff(val * 0.70710678118f));
        ((h16*)outp)[(size_t)row * Nc + c] = (h16)ge;
      } else if (EPI == 2) {
        int bw = row / 49, n = row % 49;
        int bb = bw / 64, win = bw % 64;
        int wy = win / 8, wx = win % 8;
        int ty = n / 7, tx = n % 7;
        int h = wy * 7 + ty + 3; if (h >= 56) h -= 56;   // un-shift (+3 mod 56)
        int w = wx * 7 + tx + 3; if (w >= 56) w -= 56;
        size_t idx = (((size_t)bb * 384 + c) * 56 + h) * 56 + w;
        ((float*)outp)[idx] = resid[idx] + val;
      } else {                                 // EPI == 3
        int bb = row / 3136, hw = row % 3136;
        size_t idx = ((size_t)bb * 384 + c) * 3136 + hw;
        ((float*)outp)[idx] = resid[idx] + val;
      }
    }
  }
}

// ---------------------------------------------------------------------------
// Windowed attention: one block per (window, head); 128 threads = 4 waves.
// S = (q*scale) k^T (+mask), softmax, out = P V. 49 tokens padded to 64.
// qkv rows: qkv[(bw*49+n)*1152 + {0,384,768} + head*32 + d]
// ---------------------------------------------------------------------------
__global__ __launch_bounds__(128) void attn_kernel(const h16* __restrict__ qkv,
                                                   h16* __restrict__ attn_out) {
  __shared__ float sS[64 * 68];    // scores, f32, padded stride (bank-safe)
  __shared__ h16   sP[64 * 80];    // softmax probs, f16, 16B-aligned rows
  __shared__ h16   sVT[32 * 80];   // V^T [d][token], f16
  const int tid  = threadIdx.x;
  const int lane = tid & 31;
  const int wave = tid >> 5;                  // M-tile (rows 16*wave..)
  const int bw   = blockIdx.x;                // b*64 + win
  const int head = blockIdx.y;
  const int win  = bw % 64;
  const int wy = win / 8, wx = win % 8;
  const size_t rowbase = (size_t)bw * 49;
  const int hs = lane >> 4;

  // stage V^T into LDS (zero-fill token padding 49..63)
  for (int e = tid; e < 64 * 32; e += 128) {
    int mm = e >> 5, d = e & 31;
    h16 v = (h16)0.f;
    if (mm < 49) v = qkv[(rowbase + mm) * 1152 + 768 + head * 32 + d];
    sVT[d * 80 + mm] = v;
  }

  // A fragment of q for this wave's 16 rows (clamp row for address safety)
  int mq = 16 * wave + (lane & 15); if (mq > 48) mq = 48;
  const h16* qp = qkv + (rowbase + mq) * 1152 + head * 32 + hs * 8;
  v16h qf = frag16(qp, qp + 16);

  // S = q k^T : 4 N-tiles, single K step (HD=32)
  v8f zero = {};
#pragma unroll
  for (int nt = 0; nt < 4; ++nt) {
    int mc = nt * 16 + (lane & 15); if (mc > 48) mc = 48;
    const h16* kp = qkv + (rowbase + mc) * 1152 + 384 + head * 32 + hs * 8;
    v16h kf = frag16(kp, kp + 16);
    v8f sacc = wmma_f16(qf, kf, zero);
#pragma unroll
    for (int r = 0; r < 8; ++r)
      sS[(16 * wave + r + hs * 8) * 68 + nt * 16 + (lane & 15)] = sacc[r];
  }
  __syncthreads();

  // softmax per row with analytic shifted-window mask
  if (tid < 64) {
    int rrow = tid;
    if (rrow < 49) {
      int ty = rrow / 7, tx = rrow % 7;
      int gh = wy * 7 + ty, gw = wx * 7 + tx;
      int rh = (gh < 49) ? 0 : (gh < 53 ? 1 : 2);
      int rw = (gw < 49) ? 0 : (gw < 53 ? 1 : 2);
      int reg_n = rh * 3 + rw;
      const float scale = 0.1767766952966369f;          // 1/sqrt(32)
      float mx = -1e30f;
      for (int mcol = 0; mcol < 64; ++mcol) {
        float v = -1e30f;
        if (mcol < 49) {
          int my = mcol / 7, mxx = mcol % 7;
          int g2h = wy * 7 + my, g2w = wx * 7 + mxx;
          int r2h = (g2h < 49) ? 0 : (g2h < 53 ? 1 : 2);
          int r2w = (g2w < 49) ? 0 : (g2w < 53 ? 1 : 2);
          v = sS[rrow * 68 + mcol] * scale + ((r2h * 3 + r2w == reg_n) ? 0.f : -100.f);
        }
        sS[rrow * 68 + mcol] = v;
        mx = fmaxf(mx, v);
      }
      float sum = 0.f;
      for (int mcol = 0; mcol < 64; ++mcol) {
        float e = __expf(sS[rrow * 68 + mcol] - mx);
        sS[rrow * 68 + mcol] = e;
        sum += e;
      }
      float inv = 1.f / sum;
      for (int mcol = 0; mcol < 64; ++mcol)
        sP[rrow * 80 + mcol] = (h16)(sS[rrow * 68 + mcol] * inv);
    } else {
      for (int mcol = 0; mcol < 64; ++mcol) sP[rrow * 80 + mcol] = (h16)0.f;
    }
  }
  __syncthreads();

  // out = P @ V : 2 N-tiles (d), K = 64 tokens (2 steps)
#pragma unroll
  for (int nt = 0; nt < 2; ++nt) {
    v8f oacc = zero;
#pragma unroll
    for (int kt = 0; kt < 2; ++kt) {
      const h16* pp = &sP[(16 * wave + (lane & 15)) * 80 + kt * 32 + hs * 8];
      v16h pf = frag16(pp, pp + 16);
      const h16* vp = &sVT[(nt * 16 + (lane & 15)) * 80 + kt * 32 + hs * 8];
      v16h vf = frag16(vp, vp + 16);
      oacc = wmma_f16(pf, vf, oacc);
    }
#pragma unroll
    for (int r = 0; r < 8; ++r) {
      int mm = 16 * wave + r + hs * 8;
      if (mm < 49)
        attn_out[(rowbase + mm) * 384 + head * 32 + nt * 16 + (lane & 15)] = (h16)oacc[r];
    }
  }
}

// ---------------------------------------------------------------------------
// Host-side orchestration (all launches on `stream`, graph-capture safe)
// ---------------------------------------------------------------------------
extern "C" void kernel_launch(void* const* d_in, const int* in_sizes, int n_in,
                              void* d_out, int out_size, void* d_ws, size_t ws_size,
                              hipStream_t stream) {
  (void)in_sizes; (void)n_in; (void)out_size; (void)ws_size;
  const float* x      = (const float*)d_in[0];
  const float* n1g    = (const float*)d_in[1];
  const float* n1b    = (const float*)d_in[2];
  const float* qkv_w  = (const float*)d_in[3];
  const float* qkv_b  = (const float*)d_in[4];
  const float* proj_w = (const float*)d_in[5];
  const float* proj_b = (const float*)d_in[6];
  const float* n2g    = (const float*)d_in[7];
  const float* n2b    = (const float*)d_in[8];
  const float* fc1_w  = (const float*)d_in[9];
  const float* fc1_b  = (const float*)d_in[10];
  const float* fc2_w  = (const float*)d_in[11];
  const float* fc2_b  = (const float*)d_in[12];

  char* ws = (char*)d_ws;
  // f16 weights, [N][K] transposed
  h16* qkvT = (h16*)(ws + 0);                 // 1152*384
  h16* projT = (h16*)(ws + 884736);           // 384*384
  h16* fc1T  = (h16*)(ws + 1179648);          // 1536*384
  h16* fc2T  = (h16*)(ws + 2359296);          // 384*1536
  // activation regions (reused across phases)
  h16*   R1 = (h16*)(ws + 3538944);           // qkv_out [50176][1152] / ln2_out [50176][384]
  h16*   R2 = (h16*)(ws + 119144448);         // xw / attn_out [50176][384] / h1 [50176][1536]
  float* R3 = (float*)(ws + 273285120);       // x2 (B,C,H,W) f32

  // weight conversion + transpose
  wcvt_kernel<<<(1152 * 384 + 255) / 256, 256, 0, stream>>>(qkv_w, qkvT, 384, 1152);
  wcvt_kernel<<<(384 * 384 + 255) / 256, 256, 0, stream>>>(proj_w, projT, 384, 384);
  wcvt_kernel<<<(1536 * 384 + 255) / 256, 256, 0, stream>>>(fc1_w, fc1T, 384, 1536);
  wcvt_kernel<<<(384 * 1536 + 255) / 256, 256, 0, stream>>>(fc2_w, fc2T, 1536, 384);

  // LN1 + shifted-window partition -> xw (R2)
  ln_partition_kernel<<<1568, 256, 0, stream>>>(x, n1g, n1b, R2, 0);

  // QKV GEMM: [50176][384] x [1152][384]^T -> R1 f16 (M tiles x, N groups y)
  gemm_kernel<0, 3><<<dim3(3136, 3), 256, 16 * 384 * 2, stream>>>(
      R2, qkvT, qkv_b, R1, nullptr, 384, 1152);

  // attention per (window, head) -> attn_out (R2, [row][384])
  attn_kernel<<<dim3(1024, 12), 128, 0, stream>>>(R1, R2);

  // proj GEMM + window reverse + unshift + residual(x) -> x2 (R3)
  gemm_kernel<2, 3><<<dim3(3136, 1), 256, 16 * 384 * 2, stream>>>(
      R2, projT, proj_b, R3, x, 384, 384);

  // LN2 (plain token order) -> R1
  ln_partition_kernel<<<1568, 256, 0, stream>>>(R3, n2g, n2b, R1, 1);

  // FC1 + exact GELU -> h1 (R2)
  gemm_kernel<1, 4><<<dim3(3136, 3), 256, 16 * 384 * 2, stream>>>(
      R1, fc1T, fc1_b, R2, nullptr, 384, 1536);

  // FC2 + residual(x2) -> final output (B,C,H,W) f32
  gemm_kernel<3, 3><<<dim3(3136, 1), 256, 16 * 1536 * 2, stream>>>(
      R2, fc2T, fc2_b, d_out, R3, 1536, 384);
}
